// graphDec_53506702573665
// MI455X (gfx1250) — matrix operand order repaired and checked by reading, compiled, not run on hardware
//
#include <hip/hip_runtime.h>
#include <hip/hip_bf16.h>
#include <cstddef>

typedef __attribute__((ext_vector_type(16))) __bf16 v16bf;
typedef __attribute__((ext_vector_type(8)))  __bf16 v8bf;
typedef __attribute__((ext_vector_type(8)))  float  v8f;

#define BN_EPS 1e-5f
#define NB 50      // real neighbors
#define NBP 64     // padded rows (4 WMMA M-tiles)
#define DE 128     // embed dim (8 WMMA N-tiles)

// ---------------------------------------------------------------------------
// Weight packer: f32 [K][128] row-major  ->  bf16 WMMA B-fragment order.
// Packed layout: [ktile][ntile][lane][16 bf16]  (ktile = K/32 tiles)
//   lane <16 : column n = ntile*16+lane,    elems j=0..15 are K = ktile*32 + {0..7, 16..23}
//   lane >=16: column n = ntile*16+lane-16, elems j=0..15 are K = ktile*32 + {8..15, 24..31}
// => each lane's fragment is one contiguous 32-byte chunk.
// ---------------------------------------------------------------------------
__global__ void pack_w_kernel(const float* __restrict__ w, __bf16* __restrict__ dst, int K) {
  int idx = blockIdx.x * 256 + threadIdx.x;
  if (idx >= K * 128) return;
  int k = idx >> 7;           // row (K index)
  int n = idx & 127;          // col (N index)
  int ktile = k >> 5, kin = k & 31;
  int half = (kin >> 3) & 1, j = (kin & 7) + ((kin >> 4) << 3);
  int ntile = n >> 4;
  int lane = half * 16 + (n & 15);
  size_t pos = ((size_t)((ktile * 8 + ntile) * 32 + lane)) * 16 + j;
  dst[pos] = (__bf16)w[idx];
}

// Build a 16-bit A fragment (16x32 tile) from LDS, per ISA 7.12.2:
// lane m (0..15): row m, K = 0..7 then 16..23 ; lane m+16: row m, K = 8..15 then 24..31
__device__ __forceinline__ v16bf load_a16(const __bf16* p) {
  v8bf lo = *(const v8bf*)p;          // K base + half*8 + (0..7)
  v8bf hi = *(const v8bf*)(p + 16);   // K base + half*8 + 16 + (0..7)
  v16bf a;
#pragma unroll
  for (int i = 0; i < 8; ++i) { a[i] = lo[i]; a[i + 8] = hi[i]; }
  return a;
}

// Accumulator epilogue: +bias, ReLU, cvt bf16, store 16x128 tile to LDS.
// C/D layout: VGPR v -> row v (lanes 0-15) / v+8 (lanes 16-31), col = lane&15 per ntile.
__device__ __forceinline__ void store_relu_bias(const v8f* acc, const float* __restrict__ bias,
                                                __bf16* outTile, int lane) {
  const int m = lane & 15, half = lane >> 4;
#pragma unroll
  for (int nt = 0; nt < 8; ++nt) {
    const int col = nt * 16 + m;
    const float bv = bias[col];
#pragma unroll
    for (int v = 0; v < 8; ++v) {
      float x = acc[nt][v] + bv;
      outTile[(v + half * 8) * DE + col] = (__bf16)fmaxf(x, 0.0f);
    }
  }
}

// 16xK @ Kx128 -> relu -> bf16 LDS tile.  A rows from LDS tile (stride aStride),
// with optional broadcast source (same row for all m, e.g. the user embedding)
// for the last KTB k-tiles.
template <int KTA, int KTB>
__device__ __forceinline__ void wave_gemm_lds(const __bf16* aTile, int aStride,
                                              const __bf16* aBcast,
                                              const __bf16* __restrict__ wPacked,
                                              const float* __restrict__ bias,
                                              __bf16* outTile, int lane) {
  v8f acc[8] = {};
  const int m = lane & 15, half = lane >> 4;
#pragma unroll
  for (int kt = 0; kt < KTA + KTB; ++kt) {
    const __bf16* p = (kt < KTA)
        ? (aTile + kt * 32 + m * aStride + half * 8)
        : (aBcast + (kt - KTA) * 32 + half * 8);
    v16bf a = load_a16(p);
    const __bf16* wp = wPacked + ((size_t)(kt * 8) * 32 + lane) * 16;
#pragma unroll
    for (int nt = 0; nt < 8; ++nt) {
      v16bf b = *(const v16bf*)(wp + (size_t)nt * 512);
      acc[nt] = __builtin_amdgcn_wmma_f32_16x16x32_bf16(
          false, a, false, b, (short)0, acc[nt], false, false);
    }
  }
  store_relu_bias(acc, bias, outTile, lane);
}

// ---------------------------------------------------------------------------
// Fused per-user kernel: 1 block (4 wave32) per user. Wave w owns rows
// [16w,16w+16) of the 64 padded neighbor rows through the whole GEMM chain.
// ---------------------------------------------------------------------------
__global__ __launch_bounds__(128) void fused_graphrec(
    const float* __restrict__ u2e, const float* __restrict__ i2e,
    const float* __restrict__ r2e, const float* __restrict__ t2e,
    const float* __restrict__ ne_b, const float* __restrict__ ne1_b,
    const float* __restrict__ att1_b, const float* __restrict__ att2_b,
    const float* __restrict__ att3_w, const float* __restrict__ att3_b,
    const float* __restrict__ lin_w, const float* __restrict__ lin_b,
    const float* __restrict__ w1_w, const float* __restrict__ w1_b,
    const float* __restrict__ bn1_g, const float* __restrict__ bn1_b,
    const float* __restrict__ bn1_m, const float* __restrict__ bn1_v,
    const float* __restrict__ w2_w, const float* __restrict__ w2_b,
    const float* __restrict__ bn2_g, const float* __restrict__ bn2_b,
    const float* __restrict__ bn2_m, const float* __restrict__ bn2_v,
    const float* __restrict__ w3_w, const float* __restrict__ w3_b,
    const int* __restrict__ user_idx, const int* __restrict__ item_idx,
    const int* __restrict__ rat_idx, const int* __restrict__ time_idx,
    const __bf16* __restrict__ wsPacked, float* __restrict__ out) {
  __shared__ __align__(16) __bf16 sBufA[NBP * DE];  // x1, then reused for a1
  __shared__ __align__(16) __bf16 sX[NBP * DE];     // x (needed through pooling)
  __shared__ __align__(16) __bf16 sA2[NBP * DE];    // a2 (logit input)
  __shared__ __align__(16) __bf16 sUbf[DE];
  __shared__ float sUf[DE];
  __shared__ float sLog[NBP], sAttW[NBP];
  __shared__ float sPool[DE], sComb[DE];
  __shared__ float sH1[32], sH2[16];

  const int tid = threadIdx.x;
  const int lane = tid & 31;
  const int wid = tid >> 5;
  const int rowbase = wid * 16;
  const int m = lane & 15, half = lane >> 4;
  const int b = blockIdx.x;

  const __bf16* wNe   = wsPacked;                    // 384x128 -> 49152 elems
  const __bf16* wNe1  = wsPacked + 49152;            // 128x128 -> 16384
  const __bf16* wAtt1 = wsPacked + 65536;            // 256x128 -> 32768
  const __bf16* wAtt2 = wsPacked + 98304;            // 128x128 -> 16384

  // ---- user embedding to LDS (f32 + bf16 copies) ----
  if (tid < DE) {
    float uv = u2e[(size_t)user_idx[b] * DE + tid];
    sUf[tid] = uv;
    sUbf[tid] = (__bf16)uv;
  }

  // ---- GEMM1: neigh[64x384] @ ne_w -> relu -> x1 (A gathered from HBM) ----
  {
    const int row = rowbase + m;
    const bool valid = row < NB;
    const int safe = valid ? row : 0;
    const float scale = valid ? 1.0f : 0.0f;  // zero padded rows
    const float* iP = i2e + (size_t)item_idx[b * NB + safe] * DE;
    const float* rP = r2e + (size_t)rat_idx[b * NB + safe] * DE;
    const float* tP = t2e + (size_t)time_idx[b * NB + safe] * DE;

    v8f acc[8] = {};
#pragma unroll
    for (int tb = 0; tb < 3; ++tb) {
      const float* base = (tb == 0) ? iP : ((tb == 1) ? rP : tP);
#pragma unroll
      for (int k4 = 0; k4 < 4; ++k4) {
        const float* p = base + k4 * 32 + half * 8;
        v16bf a;
#pragma unroll
        for (int j = 0; j < 8; ++j) {
          a[j]     = (__bf16)(p[j] * scale);
          a[j + 8] = (__bf16)(p[j + 16] * scale);
        }
        const int kt = tb * 4 + k4;
        const __bf16* wp = wNe + ((size_t)(kt * 8) * 32 + lane) * 16;
#pragma unroll
        for (int nt = 0; nt < 8; ++nt) {
          v16bf bf = *(const v16bf*)(wp + (size_t)nt * 512);
          acc[nt] = __builtin_amdgcn_wmma_f32_16x16x32_bf16(
              false, a, false, bf, (short)0, acc[nt], false, false);
        }
      }
    }
    store_relu_bias(acc, ne_b, sBufA + rowbase * DE, lane);
  }
  __syncthreads();

  // ---- GEMM2: x = relu(x1 @ ne1_w + b) ----
  wave_gemm_lds<4, 0>(sBufA + rowbase * DE, DE, nullptr, wNe1, ne1_b,
                      sX + rowbase * DE, lane);
  __syncthreads();

  // ---- att1: relu(concat(x, u) @ att1_w + b); K 128..255 broadcast user ----
  wave_gemm_lds<4, 4>(sX + rowbase * DE, DE, sUbf, wAtt1, att1_b,
                      sBufA + rowbase * DE, lane);
  __syncthreads();

  // ---- att2: relu(a1 @ att2_w + b) ----
  wave_gemm_lds<4, 0>(sBufA + rowbase * DE, DE, nullptr, wAtt2, att2_b,
                      sA2 + rowbase * DE, lane);
  __syncthreads();

  // ---- attention logits (mask padded rows) ----
  if (tid < NBP) {
    float s = -1e30f;
    if (tid < NB) {
      s = att3_b[0];
      for (int k = 0; k < DE; ++k) s += (float)sA2[tid * DE + k] * att3_w[k];
    }
    sLog[tid] = s;
  }
  __syncthreads();

  // ---- softmax over neighbors ----
  if (tid == 0) {
    float mx = -1e30f;
    for (int r = 0; r < NBP; ++r) mx = fmaxf(mx, sLog[r]);
    float sum = 0.0f;
    for (int r = 0; r < NBP; ++r) { float e = __expf(sLog[r] - mx); sAttW[r] = e; sum += e; }
    float inv = 1.0f / sum;
    for (int r = 0; r < NBP; ++r) sAttW[r] *= inv;
  }
  __syncthreads();

  // ---- pooled = sum_l att[l] * x[l,:] ----
  if (tid < DE) {
    float p = 0.0f;
    for (int r = 0; r < NB; ++r) p += sAttW[r] * (float)sX[r * DE + tid];
    sPool[tid] = p;
  }
  __syncthreads();

  // ---- comb = relu(concat(u, pooled) @ lin_w + lin_b) ----
  if (tid < DE) {
    float s = lin_b[tid];
    for (int k = 0; k < DE; ++k) s += sUf[k] * lin_w[k * DE + tid];
    for (int k = 0; k < DE; ++k) s += sPool[k] * lin_w[(k + DE) * DE + tid];
    sComb[tid] = fmaxf(s, 0.0f);
  }
  __syncthreads();

  // ---- head: 128->32 (BN,relu) ----
  if (tid < 32) {
    float s = w1_b[tid];
    for (int k = 0; k < DE; ++k) s += sComb[k] * w1_w[k * 32 + tid];
    s = (s - bn1_m[tid]) * __frsqrt_rn(bn1_v[tid] + BN_EPS) * bn1_g[tid] + bn1_b[tid];
    sH1[tid] = fmaxf(s, 0.0f);
  }
  __syncthreads();
  // ---- 32->16 (BN,relu) ----
  if (tid < 16) {
    float s = w2_b[tid];
    for (int k = 0; k < 32; ++k) s += sH1[k] * w2_w[k * 16 + tid];
    s = (s - bn2_m[tid]) * __frsqrt_rn(bn2_v[tid] + BN_EPS) * bn2_g[tid] + bn2_b[tid];
    sH2[tid] = fmaxf(s, 0.0f);
  }
  __syncthreads();
  // ---- 16->1 ----
  if (tid == 0) {
    float s = w3_b[0];
    for (int k = 0; k < 16; ++k) s += sH2[k] * w3_w[k];
    out[b] = s;
  }
}

extern "C" void kernel_launch(void* const* d_in, const int* in_sizes, int n_in,
                              void* d_out, int out_size, void* d_ws, size_t ws_size,
                              hipStream_t stream) {
  (void)in_sizes; (void)n_in; (void)out_size; (void)ws_size;
  const float* u2e    = (const float*)d_in[0];
  const float* i2e    = (const float*)d_in[1];
  const float* r2e    = (const float*)d_in[2];
  const float* t2e    = (const float*)d_in[3];
  const float* ne_w   = (const float*)d_in[4];
  const float* ne_b   = (const float*)d_in[5];
  const float* ne1_w  = (const float*)d_in[6];
  const float* ne1_b  = (const float*)d_in[7];
  const float* att1_w = (const float*)d_in[8];
  const float* att1_b = (const float*)d_in[9];
  const float* att2_w = (const float*)d_in[10];
  const float* att2_b = (const float*)d_in[11];
  const float* att3_w = (const float*)d_in[12];
  const float* att3_b = (const float*)d_in[13];
  const float* lin_w  = (const float*)d_in[14];
  const float* lin_b  = (const float*)d_in[15];
  const float* w1_w   = (const float*)d_in[16];
  const float* w1_b   = (const float*)d_in[17];
  const float* bn1_g  = (const float*)d_in[18];
  const float* bn1_b  = (const float*)d_in[19];
  const float* bn1_m  = (const float*)d_in[20];
  const float* bn1_v  = (const float*)d_in[21];
  const float* w2_w   = (const float*)d_in[22];
  const float* w2_b   = (const float*)d_in[23];
  const float* bn2_g  = (const float*)d_in[24];
  const float* bn2_b  = (const float*)d_in[25];
  const float* bn2_m  = (const float*)d_in[26];
  const float* bn2_v  = (const float*)d_in[27];
  const float* w3_w   = (const float*)d_in[28];
  const float* w3_b   = (const float*)d_in[29];
  const int* user_idx = (const int*)d_in[30];
  const int* item_idx = (const int*)d_in[31];
  const int* rat_idx  = (const int*)d_in[32];
  const int* time_idx = (const int*)d_in[33];

  __bf16* ws = (__bf16*)d_ws;
  // Pre-pack the four GEMM weight matrices into bf16 WMMA B-fragment order.
  pack_w_kernel<<<(49152 + 255) / 256, 256, 0, stream>>>(ne_w,   ws,         384);
  pack_w_kernel<<<(16384 + 255) / 256, 256, 0, stream>>>(ne1_w,  ws + 49152, 128);
  pack_w_kernel<<<(32768 + 255) / 256, 256, 0, stream>>>(att1_w, ws + 65536, 256);
  pack_w_kernel<<<(16384 + 255) / 256, 256, 0, stream>>>(att2_w, ws + 98304, 128);

  fused_graphrec<<<4096, 128, 0, stream>>>(
      u2e, i2e, r2e, t2e, ne_b, ne1_b, att1_b, att2_b, att3_w, att3_b,
      lin_w, lin_b, w1_w, w1_b, bn1_g, bn1_b, bn1_m, bn1_v,
      w2_w, w2_b, bn2_g, bn2_b, bn2_m, bn2_v, w3_w, w3_b,
      user_idx, item_idx, rat_idx, time_idx, ws, (float*)d_out);
}
